// MultiHeadSelfAttention_10033043604121
// MI455X (gfx1250) — matrix operand order repaired and checked by reading, not compiled, not run on hardware
//
#include <hip/hip_runtime.h>

// ---------------------------------------------------------------- types
typedef __attribute__((ext_vector_type(16))) _Float16 v16h;
typedef __attribute__((ext_vector_type(8)))  _Float16 v8h;
typedef __attribute__((ext_vector_type(4)))  _Float16 v4h;
typedef __attribute__((ext_vector_type(8)))  float    v8f;
typedef __attribute__((ext_vector_type(4)))  float    v4f;
typedef __attribute__((ext_vector_type(4)))  unsigned int v4u;
typedef __attribute__((ext_vector_type(8)))  int      v8i;
typedef __attribute__((ext_vector_type(4)))  int      v4i;

#define SEQ 4096
#define DIM 1024
#define NH  16
#define DHD 64
#define NKB (SEQ / 32)
#define LOG2E 1.44269504088896f
#define SM_SCALE (1.0f / 32.0f)   // reference divides by sqrt(D)=32, not sqrt(DH)

union V16U { v16h v; v8h h[2]; };

// ---------------- TDM availability / arity guard --------------------
#if defined(__has_builtin)
#  if __has_builtin(__builtin_amdgcn_tensor_load_to_lds) && \
      __has_builtin(__builtin_amdgcn_s_wait_tensorcnt)
#    define HAVE_TDM 1
#  endif
#endif
#ifndef HAVE_TDM
#  define HAVE_TDM 0
#endif

#if HAVE_TDM
#  if __has_include(<hip/amd_detail/amd_gfx1250_TDM.h>)
// therock/clang-23 headers present -> 6-arg builtin
#    define TDM_CALL(g0, g1, a, b, z8) \
       __builtin_amdgcn_tensor_load_to_lds((g0), (g1), (a), (b), (z8), 0)
#  else
// ROCm 7.2 / clang-22 -> 5-arg builtin
#    define TDM_CALL(g0, g1, a, b, z8) \
       __builtin_amdgcn_tensor_load_to_lds((g0), (g1), (a), (b), 0)
#  endif
#endif

__device__ __forceinline__ v8f wmma16(v16h a, v16h b, v8f c) {
  return __builtin_amdgcn_wmma_f32_16x16x32_f16(false, a, false, b, (short)0, c,
                                                false, false);
}

__device__ __forceinline__ float redmax16(float x) {
#pragma unroll
  for (int m = 8; m >= 1; m >>= 1) x = fmaxf(x, __shfl_xor(x, m, 16));
  return x;
}
__device__ __forceinline__ float redadd16(float x) {
#pragma unroll
  for (int m = 8; m >= 1; m >>= 1) x += __shfl_xor(x, m, 16);
  return x;
}

#if HAVE_TDM
// Issue one TENSOR_LOAD_TO_LDS of a 2-D tile (data_size = 8 bytes / elem).
//   lds_off : byte offset of destination inside LDS
//   ga      : global byte address of tile start
//   td0/td1 : tensor dims (elements / lines) for OOB checks
//   t0/t1   : tile dims (elements per line / #lines; t1==0 -> 1-D)
//   stride0 : elements between consecutive lines
__device__ __forceinline__ void tdm_load(unsigned lds_off,
                                         unsigned long long ga,
                                         unsigned td0, unsigned td1,
                                         unsigned t0, unsigned t1,
                                         unsigned long long stride0) {
  v4u g0;
  g0[0] = 1u;                                            // count=1, user mode
  g0[1] = lds_off;                                       // lds_addr
  g0[2] = (unsigned)ga;                                  // global_addr[31:0]
  g0[3] = ((unsigned)(ga >> 32) & 0x01FFFFFFu) | 0x80000000u;  // [56:32] | type=2
  v8i g1;
  g1[0] = (int)(3u << 16);                               // data_size = 8B
  g1[1] = (int)((td0 & 0xFFFFu) << 16);                  // tensor_dim0[15:0]
  g1[2] = (int)((td0 >> 16) | ((td1 & 0xFFFFu) << 16));  // dim0[31:16]|dim1[15:0]
  g1[3] = (int)((td1 >> 16) | (t0 << 16));               // dim1[31:16]|tile_dim0
  g1[4] = (int)t1;                                       // tile_dim1 (tile_dim2=0)
  g1[5] = (int)(unsigned)stride0;                        // dim0_stride[31:0]
  g1[6] = (int)(unsigned)(stride0 >> 32);                // dim0_stride[47:32]
  g1[7] = 0;
  v4i z4 = {};
  v8i z8 = {};
  TDM_CALL(g0, g1, z4, z4, z8);
}
#endif

// ---------------------------------------------------------------- packing
__global__ void cvt_f32_f16_k(const float* __restrict__ src,
                              _Float16* __restrict__ dst) {
  size_t i = ((size_t)blockIdx.x * blockDim.x + threadIdx.x) * 4;
  v4f v = *(const v4f*)(src + i);
  *(v4h*)(dst + i) = __builtin_convertvector(v, v4h);
}

// W[H][D][DH] fp32  ->  Wt[j = h*64+e][d] f16 (column-major for B-frags)
__global__ void pack_whead_k(const float* __restrict__ W,
                             _Float16* __restrict__ Wt) {
  int d = blockIdx.x * blockDim.x + threadIdx.x;  // 0..1023
  int j = blockIdx.y;                              // 0..1023
  int h = j >> 6, e = j & 63;
  Wt[(size_t)j * DIM + d] = (_Float16)W[((size_t)h * DIM + d) * DHD + e];
}

// Wo[d][j] fp32 -> Wt[j][d] f16
__global__ void pack_wo_k(const float* __restrict__ W,
                          _Float16* __restrict__ Wt) {
  int d = blockIdx.x * blockDim.x + threadIdx.x;
  int j = blockIdx.y;
  Wt[(size_t)j * DIM + d] = (_Float16)W[(size_t)d * DIM + j];
}

// ---------------------------------------------------------------- projection GEMM
// Out(16x64 per wave) = X[SEQ][DIM](f16) * Wt(col-major f16) + bias
// transOut==0 : Out is [H][S][DH] f16 (Q/K layout)
// transOut==1 : Out is [H*DH][S]  f16 (V transposed layout)
__global__ void proj_gemm_k(const _Float16* __restrict__ X,
                            const _Float16* __restrict__ Wt,
                            const float* __restrict__ bias,
                            _Float16* __restrict__ Out, int transOut) {
  const int lane = threadIdx.x;
  const int col  = lane & 15;
  const bool hi  = lane >= 16;
  const int o8   = hi ? 8 : 0;
  const int o16  = hi ? 16 : 0;
  const int mt   = blockIdx.x;       // 16-row tile
  const int ng   = blockIdx.y;       // group of 64 cols
  const _Float16* Xrow = X + (size_t)(mt * 16 + col) * DIM;

  v8f acc[4];
#pragma unroll
  for (int t = 0; t < 4; ++t) {
    float b = bias[ng * 64 + t * 16 + col];
#pragma unroll
    for (int r = 0; r < 8; ++r) acc[t][r] = b;
  }

  for (int k = 0; k < DIM; k += 32) {
    V16U a;
    a.h[0] = *(const v8h*)(Xrow + k + o16);
    a.h[1] = *(const v8h*)(Xrow + k + 8 + o16);
#pragma unroll
    for (int t = 0; t < 4; ++t) {
      v16h bf = *(const v16h*)(Wt + (size_t)(ng * 64 + t * 16 + col) * DIM + k + o16);
      acc[t] = wmma16(a.v, bf, acc[t]);
    }
  }

#pragma unroll
  for (int t = 0; t < 4; ++t) {
    int j = ng * 64 + t * 16 + col;
    if (transOut) {
      v8h pk;
#pragma unroll
      for (int r = 0; r < 8; ++r) pk[r] = (_Float16)acc[t][r];
      *(v8h*)(Out + (size_t)j * SEQ + mt * 16 + o8) = pk;
    } else {
#pragma unroll
      for (int r = 0; r < 8; ++r) {
        int row = mt * 16 + o8 + r;
        Out[(size_t)(j >> 6) * SEQ * DHD + (size_t)row * DHD + (j & 63)] =
            (_Float16)acc[t][r];
      }
    }
  }
}

// ---------------------------------------------------------------- flash attention
// One wave: 16 queries x one head. Online softmax over 128 key-blocks of 32.
// K/V tiles are staged into LDS by the Tensor Data Mover, double-buffered,
// synchronized with TENSORcnt so DMA of block kb+1 overlaps compute of kb.
__global__ void flash_attn_k(const _Float16* __restrict__ Qh,
                             const _Float16* __restrict__ Kh,
                             const _Float16* __restrict__ Vt,
                             _Float16* __restrict__ AO) {
  __shared__ _Float16 lds_p[16 * 32];                 // P tile [16 q][32 keys]
#if HAVE_TDM
  __shared__ __align__(128) _Float16 kbuf[2][32 * 64];  // 2 x 4KB K tiles
  __shared__ __align__(128) _Float16 vbuf[2][64 * 32];  // 2 x 4KB V tiles
#endif
  const int lane = threadIdx.x;
  const int col  = lane & 15;
  const bool hi  = lane >= 16;
  const int o8   = hi ? 8 : 0;
  const int o16  = hi ? 16 : 0;
  const int rb   = hi ? 8 : 0;          // row base for C-layout
  const int qt   = blockIdx.x;
  const int h    = blockIdx.y;
  const int s0   = qt * 16;

  // Q A-frags (dh 0..31 and 32..63), row = s0+col
  const _Float16* Qrow = Qh + ((size_t)h * SEQ + (s0 + col)) * DHD;
  V16U qa0, qa1;
  qa0.h[0] = *(const v8h*)(Qrow + 0 + o16);
  qa0.h[1] = *(const v8h*)(Qrow + 8 + o16);
  qa1.h[0] = *(const v8h*)(Qrow + 32 + o16);
  qa1.h[1] = *(const v8h*)(Qrow + 40 + o16);

  float m[8], l[8], alpha[8];
  v8f oacc[4];
#pragma unroll
  for (int r = 0; r < 8; ++r) { m[r] = -1.0e30f; l[r] = 0.0f; }
#pragma unroll
  for (int t = 0; t < 4; ++t)
#pragma unroll
    for (int r = 0; r < 8; ++r) oacc[t][r] = 0.0f;

#if HAVE_TDM
  const unsigned long long kbase = (unsigned long long)(size_t)Kh;
  const unsigned long long vbase = (unsigned long long)(size_t)Vt;
  // Issue the DMA pair for a key block into buffer b.
  auto issue = [&](int kb, int b) {
    // K tile: contiguous 32 rows x 128B = 4KB (1-D, 512 x 8B elems)
    unsigned long long kga =
        kbase + ((unsigned long long)h * SEQ + (unsigned)kb * 32u) * DHD * 2u;
    tdm_load((unsigned)(size_t)(void*)&kbuf[b][0], kga,
             /*td0=*/512u, /*td1=*/1u, /*t0=*/512u, /*t1=*/0u, /*stride0=*/512u);
    // V tile: 64 lines x 64B, line stride SEQ*2 bytes = 1024 x 8B elems
    unsigned long long vga =
        vbase + ((unsigned long long)h * DHD * SEQ + (unsigned)kb * 32u) * 2u;
    tdm_load((unsigned)(size_t)(void*)&vbuf[b][0], vga,
             /*td0=*/1024u, /*td1=*/1024u, /*t0=*/8u, /*t1=*/64u,
             /*stride0=*/1024u);
  };
  issue(0, 0);
#endif

  for (int kb = 0; kb < NKB; ++kb) {
#if HAVE_TDM
    if (kb + 1 < NKB) {
      issue(kb + 1, (kb + 1) & 1);
      // in-order TDM completion: <=2 outstanding => current pair landed
      __builtin_amdgcn_s_wait_tensorcnt(2);
    } else {
      __builtin_amdgcn_s_wait_tensorcnt(0);
    }
    asm volatile("" ::: "memory");
    const _Float16* Ksrc = &kbuf[kb & 1][0];
    const _Float16* Vsrc = &vbuf[kb & 1][0];
#endif

    // ---- scores: two 16x16 tiles, K-dim = DH = 64 (2 wmma each)
    v8f st[2];
#pragma unroll
    for (int kt = 0; kt < 2; ++kt) {
#if HAVE_TDM
      const _Float16* Krow = Ksrc + (kt * 16 + col) * DHD;
#else
      const _Float16* Krow =
          Kh + ((size_t)h * SEQ + (kb * 32 + kt * 16 + col)) * DHD;
#endif
      v16h kf0 = *(const v16h*)(Krow + 0 + o16);
      v16h kf1 = *(const v16h*)(Krow + 32 + o16);
      v8f z;
#pragma unroll
      for (int r = 0; r < 8; ++r) z[r] = 0.0f;
      st[kt] = wmma16(qa0.v, kf0, z);
      st[kt] = wmma16(qa1.v, kf1, st[kt]);
    }

    // ---- online softmax (per-row state lives in all 16 lanes of the group)
#pragma unroll
    for (int r = 0; r < 8; ++r) {
      float t0 = st[0][r] * SM_SCALE;
      float t1 = st[1][r] * SM_SCALE;
      float bm = redmax16(fmaxf(t0, t1));
      float mn = fmaxf(m[r], bm);
      alpha[r] = exp2f((m[r] - mn) * LOG2E);
      float p0 = exp2f((t0 - mn) * LOG2E);
      float p1 = exp2f((t1 - mn) * LOG2E);
      m[r] = mn;
      l[r] = l[r] * alpha[r] + redadd16(p0 + p1);
      lds_p[(rb + r) * 32 + col]      = (_Float16)p0;
      lds_p[(rb + r) * 32 + 16 + col] = (_Float16)p1;
    }
    // single-wave LDS is in-order; fence compiler + DScnt before A-frag reload
    asm volatile("s_wait_dscnt 0" ::: "memory");

    // ---- P in A-frag layout (16 x 32 keys)
    V16U pa;
    pa.h[0] = *(const v8h*)(lds_p + col * 32 + o16);
    pa.h[1] = *(const v8h*)(lds_p + col * 32 + 8 + o16);

    // ---- O = diag(alpha)*O + P x V
#pragma unroll
    for (int t = 0; t < 4; ++t) {
#pragma unroll
      for (int r = 0; r < 8; ++r) oacc[t][r] *= alpha[r];
#if HAVE_TDM
      v16h vf = *(const v16h*)(Vsrc + (t * 16 + col) * 32 + o16);
#else
      v16h vf = *(const v16h*)(Vt + ((size_t)h * DHD + t * 16 + col) * SEQ +
                               kb * 32 + o16);
#endif
      oacc[t] = wmma16(pa.v, vf, oacc[t]);
    }
    asm volatile("" ::: "memory");  // keep next iter's LDS stores after the reads
  }

  // ---- finalize: divide by l, write concat layout [S][H*DH] f16
  float inv[8];
#pragma unroll
  for (int r = 0; r < 8; ++r) inv[r] = 1.0f / l[r];
#pragma unroll
  for (int t = 0; t < 4; ++t)
#pragma unroll
    for (int r = 0; r < 8; ++r) {
      int row = s0 + rb + r;
      int j = h * DHD + t * 16 + col;
      AO[(size_t)row * DIM + j] = (_Float16)(oacc[t][r] * inv[r]);
    }
}

// ---------------------------------------------------------------- output GEMM (f32 out)
__global__ void out_gemm_k(const _Float16* __restrict__ X,
                           const _Float16* __restrict__ Wt,
                           const float* __restrict__ bias,
                           float* __restrict__ Out) {
  const int lane = threadIdx.x;
  const int col  = lane & 15;
  const bool hi  = lane >= 16;
  const int o8   = hi ? 8 : 0;
  const int o16  = hi ? 16 : 0;
  const int mt   = blockIdx.x;
  const int ng   = blockIdx.y;
  const _Float16* Xrow = X + (size_t)(mt * 16 + col) * DIM;

  v8f acc[4];
#pragma unroll
  for (int t = 0; t < 4; ++t) {
    float b = bias[ng * 64 + t * 16 + col];
#pragma unroll
    for (int r = 0; r < 8; ++r) acc[t][r] = b;
  }

  for (int k = 0; k < DIM; k += 32) {
    V16U a;
    a.h[0] = *(const v8h*)(Xrow + k + o16);
    a.h[1] = *(const v8h*)(Xrow + k + 8 + o16);
#pragma unroll
    for (int t = 0; t < 4; ++t) {
      v16h bf = *(const v16h*)(Wt + (size_t)(ng * 64 + t * 16 + col) * DIM + k + o16);
      acc[t] = wmma16(a.v, bf, acc[t]);
    }
  }

#pragma unroll
  for (int t = 0; t < 4; ++t) {
    int j = ng * 64 + t * 16 + col;
#pragma unroll
    for (int r = 0; r < 8; ++r) {
      int row = mt * 16 + o8 + r;
      Out[(size_t)row * DIM + j] = acc[t][r];
    }
  }
}

// ---------------------------------------------------------------- launch
extern "C" void kernel_launch(void* const* d_in, const int* in_sizes, int n_in,
                              void* d_out, int out_size, void* d_ws, size_t ws_size,
                              hipStream_t stream) {
  const float* q  = (const float*)d_in[0];
  const float* k  = (const float*)d_in[1];
  const float* v  = (const float*)d_in[2];
  const float* Wq = (const float*)d_in[3];
  const float* bq = (const float*)d_in[4];
  const float* Wk = (const float*)d_in[5];
  const float* bk = (const float*)d_in[6];
  const float* Wv = (const float*)d_in[7];
  const float* bv = (const float*)d_in[8];
  const float* Wo = (const float*)d_in[9];
  const float* bo = (const float*)d_in[10];

  char* ws = (char*)d_ws;
  const size_t MB = 1024 * 1024;
  _Float16* xq  = (_Float16*)(ws + 0 * MB);    // [S][D] f16
  _Float16* xk  = (_Float16*)(ws + 8 * MB);
  _Float16* xv  = (_Float16*)(ws + 16 * MB);
  _Float16* WtQ = (_Float16*)(ws + 24 * MB);   // [1024][1024] col-major f16
  _Float16* WtK = (_Float16*)(ws + 26 * MB);
  _Float16* WtV = (_Float16*)(ws + 28 * MB);
  _Float16* WtO = (_Float16*)(ws + 30 * MB);
  _Float16* Qh  = (_Float16*)(ws + 32 * MB);   // [H][S][DH]
  _Float16* Kh  = (_Float16*)(ws + 40 * MB);   // [H][S][DH]
  _Float16* Vt  = (_Float16*)(ws + 48 * MB);   // [H*DH][S]
  _Float16* AO  = (_Float16*)(ws + 56 * MB);   // [S][H*DH]

  // 1) inputs fp32 -> f16
  dim3 cgrid(SEQ * DIM / 4 / 256), cblk(256);
  cvt_f32_f16_k<<<cgrid, cblk, 0, stream>>>(q, xq);
  cvt_f32_f16_k<<<cgrid, cblk, 0, stream>>>(k, xk);
  cvt_f32_f16_k<<<cgrid, cblk, 0, stream>>>(v, xv);

  // 2) weights -> col-major f16
  dim3 pgrid(DIM / 256, DIM), pblk(256);
  pack_whead_k<<<pgrid, pblk, 0, stream>>>(Wq, WtQ);
  pack_whead_k<<<pgrid, pblk, 0, stream>>>(Wk, WtK);
  pack_whead_k<<<pgrid, pblk, 0, stream>>>(Wv, WtV);
  pack_wo_k  <<<pgrid, pblk, 0, stream>>>(Wo, WtO);

  // 3) QKV projections (WMMA), V written transposed
  dim3 ggrid(SEQ / 16, DIM / 64), gblk(32);
  proj_gemm_k<<<ggrid, gblk, 0, stream>>>(xq, WtQ, bq, Qh, 0);
  proj_gemm_k<<<ggrid, gblk, 0, stream>>>(xk, WtK, bk, Kh, 0);
  proj_gemm_k<<<ggrid, gblk, 0, stream>>>(xv, WtV, bv, Vt, 1);

  // 4) flash attention: one wave per (16-query tile, head), TDM-staged K/V
  dim3 agrid(SEQ / 16, NH), ablk(32);
  flash_attn_k<<<agrid, ablk, 0, stream>>>(Qh, Kh, Vt, AO);

  // 5) output projection (f32 result)
  out_gemm_k<<<ggrid, gblk, 0, stream>>>(AO, WtO, bo, (float*)d_out);
}
